// GATLayer_53163105190236
// MI455X (gfx1250) — matrix operand order repaired and checked by reading, not compiled, not run on hardware
//
#include <hip/hip_runtime.h>

// ---------------------------------------------------------------------------
// GAT layer for MI455X (gfx1250, wave32, WMMA).
//   B=4, N=2048, IN=256, H=4, D=64.
// Strategy: bf16 WMMA (v_wmma_f32_16x16x32_bf16) for both GEMMs, flash-style
// fused attention so the [B,H,N,N] logits tensor is never materialized.
// ---------------------------------------------------------------------------

typedef __attribute__((ext_vector_type(16))) __bf16 v16bf;
typedef __attribute__((ext_vector_type(8)))  __bf16 v8bf;
typedef __attribute__((ext_vector_type(8)))  float  v8f;

constexpr int B_  = 4;
constexpr int N_  = 2048;
constexpr int IN_ = 256;
constexpr int H_  = 4;
constexpr int D_  = 64;
constexpr int HD_ = H_ * D_;   // 256

// workspace byte offsets
constexpr size_t XBF_OFF = 0;                          // 8192*256 bf16 = 4 MiB
constexpr size_t WT_OFF  = XBF_OFF + (size_t)B_*N_*IN_*2;      // 256*256 bf16
constexpr size_t HT_OFF  = WT_OFF  + (size_t)IN_*HD_*2;        // 4*4*64*2048 bf16 = 4 MiB
constexpr size_t SRC_OFF = HT_OFF  + (size_t)B_*H_*D_*N_*2;    // 32768 f32
constexpr size_t DST_OFF = SRC_OFF + (size_t)B_*H_*N_*4;       // 32768 f32

// ---------------------------------------------------------------------------
// x (f32) -> xbf (bf16), row-major [B*N, IN]
__global__ void k_convert_x(const float* __restrict__ x, __bf16* __restrict__ xbf) {
    int i = blockIdx.x * 256 + threadIdx.x;       // exactly B*N*IN threads
    xbf[i] = (__bf16)x[i];
}

// W [IN, HD] f32 -> Wt [HD, IN] bf16 (transposed so B-tiles load contiguously)
__global__ void k_convert_w(const float* __restrict__ W, __bf16* __restrict__ Wt) {
    int i = blockIdx.x * 256 + threadIdx.x;       // exactly HD*IN threads
    int n = i >> 8;                               // output channel
    int k = i & 255;                              // input channel
    Wt[i] = (__bf16)W[k * HD_ + n];               // Wt[n*IN + k]
}

// ---------------------------------------------------------------------------
// h = x @ W via WMMA bf16. One wave per 16x16 output tile, K-loop of 8.
// Writes h transposed+bf16: hT[((b*H + h)*D + d) * N + n]  (n contiguous).
__global__ void k_gemm_h(const __bf16* __restrict__ xbf,
                         const __bf16* __restrict__ Wt,
                         __bf16* __restrict__ hT) {
    const int lane  = threadIdx.x & 31;
    const int wave  = blockIdx.x * (blockDim.x >> 5) + (threadIdx.x >> 5);
    const int mtile = wave >> 4;                  // 0..511  (rows of [B*N])
    const int ntile = wave & 15;                  // 0..15   (cols of [HD])
    const int row   = lane & 15;                  // A row / B col / C col
    const int half  = lane >> 4;

    const __bf16* arow = xbf + (size_t)(mtile * 16 + row) * IN_;
    const __bf16* brow = Wt  + (size_t)(ntile * 16 + row) * IN_;

    v8f acc = {};
    #pragma unroll
    for (int kb = 0; kb < 8; ++kb) {
        // A 16x32 bf16 layout: lane holds K runs [half*8 .. +7] and [16+half*8 .. +7]
        v8bf alo = *(const v8bf*)(arow + kb * 32 + half * 8);
        v8bf ahi = *(const v8bf*)(arow + kb * 32 + 16 + half * 8);
        // B 32x16 bf16 layout: lane holds K runs [half*8 .. +7] and [16+half*8 .. +7]
        v8bf blo = *(const v8bf*)(brow + kb * 32 + half * 8);
        v8bf bhi = *(const v8bf*)(brow + kb * 32 + 16 + half * 8);
        v16bf a, b;
        #pragma unroll
        for (int e = 0; e < 8; ++e) {
            a[e] = alo[e]; a[e + 8] = ahi[e];
            b[e] = blo[e]; b[e + 8] = bhi[e];
        }
        acc = __builtin_amdgcn_wmma_f32_16x16x32_bf16(
            false, a, false, b, (short)0, acc, false, false);
    }

    // C layout: lane holds rows M = half*8 + v (v = vgpr idx), col = row.
    const int gcol = ntile * 16 + row;            // channel in [0,256)
    const int hh   = gcol >> 6;                   // head
    const int d    = gcol & 63;
    const int bb   = mtile >> 7;                  // batch
    const int nb   = (mtile & 127) * 16 + half * 8;  // node base (8 consecutive)
    v8bf pack;
    #pragma unroll
    for (int v = 0; v < 8; ++v) pack[v] = (__bf16)acc[v];
    *(v8bf*)(hT + ((size_t)(bb * H_ + hh) * D_ + d) * N_ + nb) = pack;
}

// ---------------------------------------------------------------------------
// src[b,h,n] = sum_d hT[b,h,d,n] * a_src[h,d]  (and same for dst).
// Stored [b][h][n] so the flash kernel reads dst_j contiguously.
__global__ void k_scores(const __bf16* __restrict__ hT,
                         const float* __restrict__ a_src,
                         const float* __restrict__ a_dst,
                         float* __restrict__ srcBH,
                         float* __restrict__ dstBH) {
    int t = blockIdx.x * 256 + threadIdx.x;       // exactly B*H*N threads
    int n = t & (N_ - 1);
    int h = (t >> 11) & (H_ - 1);
    int b = t >> 13;
    const __bf16* base = hT + (size_t)(b * H_ + h) * D_ * N_ + n;
    float ss = 0.f, sd = 0.f;
    #pragma unroll 4
    for (int d = 0; d < D_; ++d) {
        float hv = (float)base[(size_t)d * N_];
        ss += hv * a_src[h * D_ + d];
        sd += hv * a_dst[h * D_ + d];
    }
    srcBH[t] = ss;
    dstBH[t] = sd;
}

// ---------------------------------------------------------------------------
// Fused flash-GAT: one wave per (b, h, 16-row i-tile). Online softmax over
// j-blocks of 32; P tile built directly in WMMA A-layout; acc via 4 bf16 WMMAs.
__global__ void k_attn(const __bf16* __restrict__ hT,
                       const float* __restrict__ srcBH,
                       const float* __restrict__ dstBH,
                       const int*   __restrict__ A_mask,
                       float* __restrict__ out) {
    const int lane  = threadIdx.x & 31;
    const int wave  = blockIdx.x * (blockDim.x >> 5) + (threadIdx.x >> 5); // 0..2047
    const int itile = wave & 127;
    const int h     = (wave >> 7) & (H_ - 1);
    const int b     = wave >> 9;
    const int row   = lane & 15;
    const int half  = lane >> 4;
    const int i0    = itile * 16;

    const float  s_i   = srcBH[(size_t)(b * H_ + h) * N_ + i0 + row];
    const float* dbase = dstBH + (size_t)(b * H_ + h) * N_;
    const int*   mbase = A_mask + (size_t)(i0 + row) * N_;
    const __bf16* vbase = hT + (size_t)(b * H_ + h) * D_ * N_;

    v8f acc0 = {}, acc1 = {}, acc2 = {}, acc3 = {};
    float m_run = -1e30f, l_run = 0.f;

    for (int j0 = 0; j0 < N_; j0 += 32) {
        // lane's 16 j-offsets: [half*8 .. +7] and [16+half*8 .. +7] (A-layout)
        float dv[16]; int mv[16];
        {
            const float4* dp0 = (const float4*)(dbase + j0 + half * 8);
            const float4* dp1 = (const float4*)(dbase + j0 + 16 + half * 8);
            float4 a0 = dp0[0], a1 = dp0[1], a2 = dp1[0], a3 = dp1[1];
            dv[0]=a0.x; dv[1]=a0.y; dv[2]=a0.z;  dv[3]=a0.w;
            dv[4]=a1.x; dv[5]=a1.y; dv[6]=a1.z;  dv[7]=a1.w;
            dv[8]=a2.x; dv[9]=a2.y; dv[10]=a2.z; dv[11]=a2.w;
            dv[12]=a3.x; dv[13]=a3.y; dv[14]=a3.z; dv[15]=a3.w;
            const int4* mp0 = (const int4*)(mbase + j0 + half * 8);
            const int4* mp1 = (const int4*)(mbase + j0 + 16 + half * 8);
            int4 b0 = mp0[0], b1 = mp0[1], b2 = mp1[0], b3 = mp1[1];
            mv[0]=b0.x; mv[1]=b0.y; mv[2]=b0.z;  mv[3]=b0.w;
            mv[4]=b1.x; mv[5]=b1.y; mv[6]=b1.z;  mv[7]=b1.w;
            mv[8]=b2.x; mv[9]=b2.y; mv[10]=b2.z; mv[11]=b2.w;
            mv[12]=b3.x; mv[13]=b3.y; mv[14]=b3.z; mv[15]=b3.w;
        }

        // logits = leaky_relu(src_i + dst_j), masked -> -1e30 sentinel
        float lg[16];
        float mx = -1e30f;
        #pragma unroll
        for (int e = 0; e < 16; ++e) {
            float v = s_i + dv[e];
            v = (v >= 0.f) ? v : 0.2f * v;
            if (mv[e] == 0) v = -1e30f;
            lg[e] = v;
            mx = fmaxf(mx, v);
        }
        // row stats live in lane and lane^16
        mx = fmaxf(mx, __shfl_xor(mx, 16, 32));
        float m_new = fmaxf(m_run, mx);
        float corr  = __expf(m_run - m_new);

        float psum = 0.f;
        v16bf pa;
        #pragma unroll
        for (int e = 0; e < 16; ++e) {
            float p = __expf(lg[e] - m_new);
            psum += p;
            pa[e] = (__bf16)p;
        }
        psum += __shfl_xor(psum, 16, 32);
        l_run = l_run * corr + psum;
        m_run = m_new;

        // rescale accumulators: C vgpr v holds row (half*8 + v); corr for row r
        // is held by lane r.
        #pragma unroll
        for (int v = 0; v < 8; ++v) {
            float cv = __shfl(corr, half * 8 + v, 32);
            acc0[v] *= cv; acc1[v] *= cv; acc2[v] *= cv; acc3[v] *= cv;
        }

        // V tiles: B[k][col] = hT[.., d = nt*16+col, n = j0 + k], k runs [half*8..+7], [16+half*8..+7]
        const __bf16* vb = vbase + j0 + half * 8;
        #pragma unroll
        for (int nt = 0; nt < 4; ++nt) {
            const __bf16* p = vb + (size_t)(nt * 16 + row) * N_;
            v8bf lo = *(const v8bf*)p;
            v8bf hi = *(const v8bf*)(p + 16);
            v16bf bm;
            #pragma unroll
            for (int e = 0; e < 8; ++e) { bm[e] = lo[e]; bm[e + 8] = hi[e]; }
            if (nt == 0) acc0 = __builtin_amdgcn_wmma_f32_16x16x32_bf16(false, pa, false, bm, (short)0, acc0, false, false);
            if (nt == 1) acc1 = __builtin_amdgcn_wmma_f32_16x16x32_bf16(false, pa, false, bm, (short)0, acc1, false, false);
            if (nt == 2) acc2 = __builtin_amdgcn_wmma_f32_16x16x32_bf16(false, pa, false, bm, (short)0, acc2, false, false);
            if (nt == 3) acc3 = __builtin_amdgcn_wmma_f32_16x16x32_bf16(false, pa, false, bm, (short)0, acc3, false, false);
        }
    }

    // epilogue: divide by row sum, ELU, store fp32 out[b, n, h*64 + d]
    const int col = row;
    #pragma unroll
    for (int v = 0; v < 8; ++v) {
        float li   = __shfl(l_run, half * 8 + v, 32);
        float linv = 1.0f / li;
        int n = i0 + half * 8 + v;
        size_t obase = (size_t)(b * N_ + n) * HD_ + h * D_;
        float o;
        o = acc0[v] * linv; o = (o > 0.f) ? o : (__expf(o) - 1.f); out[obase +  0 + col] = o;
        o = acc1[v] * linv; o = (o > 0.f) ? o : (__expf(o) - 1.f); out[obase + 16 + col] = o;
        o = acc2[v] * linv; o = (o > 0.f) ? o : (__expf(o) - 1.f); out[obase + 32 + col] = o;
        o = acc3[v] * linv; o = (o > 0.f) ? o : (__expf(o) - 1.f); out[obase + 48 + col] = o;
    }
}

// ---------------------------------------------------------------------------
extern "C" void kernel_launch(void* const* d_in, const int* in_sizes, int n_in,
                              void* d_out, int out_size, void* d_ws, size_t ws_size,
                              hipStream_t stream) {
    (void)in_sizes; (void)n_in; (void)out_size; (void)ws_size;
    const float* x      = (const float*)d_in[0];
    const int*   A_mask = (const int*)  d_in[1];
    const float* W      = (const float*)d_in[2];
    const float* a_src  = (const float*)d_in[3];
    const float* a_dst  = (const float*)d_in[4];
    float* out = (float*)d_out;

    char* ws = (char*)d_ws;
    __bf16* xbf   = (__bf16*)(ws + XBF_OFF);
    __bf16* Wt    = (__bf16*)(ws + WT_OFF);
    __bf16* hT    = (__bf16*)(ws + HT_OFF);
    float*  srcBH = (float*) (ws + SRC_OFF);
    float*  dstBH = (float*) (ws + DST_OFF);

    k_convert_x<<<(B_ * N_ * IN_) / 256, 256, 0, stream>>>(x, xbf);
    k_convert_w<<<(IN_ * HD_) / 256, 256, 0, stream>>>(W, Wt);
    // 512 m-tiles * 16 n-tiles = 8192 waves, 4 waves/block
    k_gemm_h<<<2048, 128, 0, stream>>>(xbf, Wt, hT);
    k_scores<<<(B_ * H_ * N_) / 256, 256, 0, stream>>>(hT, a_src, a_dst, srcBH, dstBH);
    // B*H*(N/16) = 2048 waves, 4 waves/block
    k_attn<<<512, 128, 0, stream>>>(hT, srcBH, dstBH, A_mask, out);
}
